// SelfAttentionBlock_60017872994684
// MI455X (gfx1250) — matrix operand order repaired and checked by reading, not compiled, not run on hardware
//
#include <hip/hip_runtime.h>

#define BB 4
#define CC 512
#define DD 64
#define NN 4096
#define GG 32
#define GC 16
#define EPSV 1e-5f

typedef _Float16 half_t;
typedef __attribute__((ext_vector_type(16))) _Float16 v16h;
typedef __attribute__((ext_vector_type(8)))  _Float16 v8h;
typedef __attribute__((ext_vector_type(8)))  float    v8f;

// ---- WMMA fragment loaders (gfx1250 wave32 layouts, CDNA5 ISA) ----
// A operand: 16x32 f16, row-major source with leading dim ld (elements).
// lane<16: row=lane, K {0..7,16..23}; lane>=16: row=lane-16, K {8..15,24..31}
__device__ __forceinline__ v16h load_fragA(const half_t* p, int ld) {
  int lane = threadIdx.x & 31;
  int row  = lane & 15;
  int koff = (lane >> 4) << 3; // 0 or 8
  const half_t* r = p + row * ld + koff;
  v8h lo = *(const v8h*)(r);
  v8h hi = *(const v8h*)(r + 16);
  v16h o;
#pragma unroll
  for (int i = 0; i < 8; ++i) { o[i] = lo[i]; o[8 + i] = hi[i]; }
  return o;
}

// B operand: 32x16 f16, loaded from B^T (Nout x K) row-major with leading dim ld.
// lane<16: col=lane, K {0..7,16..23}; lane>=16: col=lane-16, K {8..15,24..31}
__device__ __forceinline__ v16h load_fragB(const half_t* bt, int ld) {
  int lane = threadIdx.x & 31;
  const half_t* r = bt + (lane & 15) * ld + ((lane >> 4) << 3);
  v8h lo = *(const v8h*)(r);
  v8h hi = *(const v8h*)(r + 16);
  v16h o;
#pragma unroll
  for (int i = 0; i < 8; ++i) { o[i] = lo[i]; o[8 + i] = hi[i]; }
  return o;
}

__device__ __forceinline__ v8f wmma_f16(v16h a, v16h b, v8f c) {
  return __builtin_amdgcn_wmma_f32_16x16x32_f16(false, a, false, b, (short)0, c, false, false);
}

// ---------------- helper kernels ----------------
__global__ __launch_bounds__(256) void k_f32_to_f16(const float* src, half_t* dst, int n) {
  int i = blockIdx.x * 256 + threadIdx.x;
  if (i < n) dst[i] = (half_t)src[i];
}

__global__ __launch_bounds__(256) void k_gn_stats(const float* x, float* mean, float* rstd) {
  __shared__ float ss[256], sq[256];
  int bg = blockIdx.x;
  int b = bg / GG, g = bg % GG;
  const float* base = x + ((size_t)b * CC + g * GC) * NN;
  float s = 0.f, q = 0.f;
  for (int i = threadIdx.x; i < GC * NN; i += 256) {
    float v = base[i];
    s += v; q += v * v;
  }
  ss[threadIdx.x] = s; sq[threadIdx.x] = q;
  __syncthreads();
  for (int off = 128; off > 0; off >>= 1) {
    if (threadIdx.x < off) {
      ss[threadIdx.x] += ss[threadIdx.x + off];
      sq[threadIdx.x] += sq[threadIdx.x + off];
    }
    __syncthreads();
  }
  if (threadIdx.x == 0) {
    float cnt = (float)(GC * NN);
    float m = ss[0] / cnt;
    float var = sq[0] / cnt - m * m;
    mean[bg] = m;
    rstd[bg] = rsqrtf(var + EPSV);
  }
}

__global__ __launch_bounds__(256) void k_gn_apply(const float* x, const float* gamma, const float* beta,
                                                  const float* mean, const float* rstd, half_t* xnT) {
  int bc = blockIdx.x;
  int b = bc / CC, c = bc % CC;
  int g = c / GC;
  float m = mean[b * GG + g], r = rstd[b * GG + g];
  float ga = gamma[c] * r;
  float be = beta[c] - m * ga;
  const float* xp = x + ((size_t)b * CC + c) * NN;
  half_t* op = xnT + (size_t)b * NN * CC + c;
  for (int n = threadIdx.x; n < NN; n += 256)
    op[(size_t)n * CC] = (half_t)(xp[n] * ga + be);
}

// ---------------- Q/K projection: one wave -> one 16(n) x 16(d) tile ----------------
__global__ __launch_bounds__(128) void k_proj_qk(const half_t* xnT, const half_t* wqh, const half_t* wkh,
                                                 const float* bq, const float* bk, half_t* Qh, half_t* Kh) {
  int wid = (blockIdx.x * 128 + threadIdx.x) >> 5;
  int dt = wid & 3;
  int nt = (wid >> 2) & 255;
  int b  = wid >> 10;
  int lane = threadIdx.x & 31;
  v8f accq, acck;
#pragma unroll
  for (int i = 0; i < 8; ++i) { accq[i] = 0.f; acck[i] = 0.f; }
  const half_t* abase = xnT + ((size_t)b * NN + nt * 16) * CC;
  const half_t* bqb = wqh + dt * 16 * CC;
  const half_t* bkb = wkh + dt * 16 * CC;
  for (int kt = 0; kt < CC / 32; ++kt) {
    v16h a  = load_fragA(abase + kt * 32, CC);
    v16h b0 = load_fragB(bqb + kt * 32, CC);
    accq = wmma_f16(a, b0, accq);
    v16h b1 = load_fragB(bkb + kt * 32, CC);
    acck = wmma_f16(a, b1, acck);
  }
  int col = lane & 15;
  int rowadd = (lane >> 4) << 3;
  int d = dt * 16 + col;
  float qb = bq[d], kb = bk[d];
#pragma unroll
  for (int r = 0; r < 8; ++r) {
    int n = nt * 16 + rowadd + r;
    size_t oidx = ((size_t)b * NN + n) * DD + d;
    Qh[oidx] = (half_t)((accq[r] + qb) * 0.125f); // fold d^-0.5 into Q
    Kh[oidx] = (half_t)(acck[r] + kb);
  }
}

// ---------------- V projection: one wave -> one 16(d) x 16(n) tile, stores V^T [B,D,N] ----------------
__global__ __launch_bounds__(128) void k_proj_v(const half_t* xnT, const half_t* wvh, const float* bv, half_t* Vt) {
  int wid = (blockIdx.x * 128 + threadIdx.x) >> 5;
  int nt = wid & 255;
  int dt = (wid >> 8) & 3;
  int b  = wid >> 10;
  int lane = threadIdx.x & 31;
  v8f acc;
#pragma unroll
  for (int i = 0; i < 8; ++i) acc[i] = 0.f;
  const half_t* abase = wvh + dt * 16 * CC;
  const half_t* bbase = xnT + ((size_t)b * NN + nt * 16) * CC;
  for (int kt = 0; kt < CC / 32; ++kt) {
    v16h a  = load_fragA(abase + kt * 32, CC);
    v16h bb = load_fragB(bbase + kt * 32, CC);
    acc = wmma_f16(a, bb, acc);
  }
  int col = lane & 15;
  int rowadd = (lane >> 4) << 3;
  int n = nt * 16 + col;
#pragma unroll
  for (int r = 0; r < 8; ++r) {
    int d = dt * 16 + rowadd + r;
    Vt[((size_t)b * DD + d) * NN + n] = (half_t)(acc[r] + bv[d]);
  }
}

// ---------------- flash attention ----------------
// Block = 4 waves, all on the same batch b; K/V 64-key blocks staged in LDS via
// double-buffered global_load_async_to_lds_b128 (ASYNCcnt) and shared by the 4 waves.
__global__ __launch_bounds__(128) void k_attn(const half_t* Qh, const half_t* Kh, const half_t* Vt, half_t* Oh) {
  __shared__ __align__(16) half_t Kbuf[2][64][64]; // [buf][key][d]   16 KB
  __shared__ __align__(16) half_t Vbuf[2][64][64]; // [buf][d][key]   16 KB
  __shared__ __align__(16) half_t Plds[4][16][64]; // per-wave P tile  8 KB
  int warp = threadIdx.x >> 5;
  int lane = threadIdx.x & 31;
  int b  = blockIdx.x >> 6;                       // 64 blocks per batch
  int q0 = (((blockIdx.x & 63) << 2) + warp) << 4;

  v16h qf[2];
#pragma unroll
  for (int s = 0; s < 2; ++s)
    qf[s] = load_fragA(Qh + ((size_t)b * NN + q0) * DD + s * 32, DD);

  v8f o[4];
  float m_i[8], l_i[8];
#pragma unroll
  for (int j = 0; j < 4; ++j)
#pragma unroll
    for (int i = 0; i < 8; ++i) o[j][i] = 0.f;
#pragma unroll
  for (int r = 0; r < 8; ++r) { m_i[r] = -1e30f; l_i[r] = 0.f; }

  half_t* pl = &Plds[warp][0][0];
  int colbase = lane & 15;
  int rowadd  = (lane >> 4) << 3;

  // each wave async-copies its quarter of the K block (contiguous 8 KB) and of the
  // V^T block (16 rows x 128 B, row stride NN*2) into LDS
  auto prefetch = [&](int blk, int buf) {
    int k0 = blk * 64;
    const char* ksrc = (const char*)(Kh + ((size_t)b * NN + k0) * DD) + warp * 2048 + lane * 16;
    unsigned kdst = (unsigned)(size_t)(&Kbuf[buf][0][0]) + warp * 2048 + lane * 16;
#pragma unroll
    for (int i = 0; i < 4; ++i)
      asm volatile("global_load_async_to_lds_b128 %0, %1, off"
                   :: "v"(kdst + i * 512),
                      "v"((unsigned long long)(size_t)(ksrc + i * 512))
                   : "memory");
    int rowin = lane >> 3, seg = lane & 7;
#pragma unroll
    for (int i = 0; i < 4; ++i) {
      int d = warp * 16 + i * 4 + rowin;
      const char* vsrc = (const char*)(Vt + ((size_t)b * DD + d) * NN + k0) + seg * 16;
      unsigned vdst = (unsigned)(size_t)(&Vbuf[buf][0][0]) + d * 128 + seg * 16;
      asm volatile("global_load_async_to_lds_b128 %0, %1, off"
                   :: "v"(vdst),
                      "v"((unsigned long long)(size_t)vsrc)
                   : "memory");
    }
  };

  prefetch(0, 0);
  for (int blk = 0; blk < NN / 64; ++blk) {
    int p = blk & 1;
    asm volatile("s_wait_asynccnt 0x0" ::: "memory"); // my async chunk for buf p landed
    __syncthreads();                                  // everyone's chunk landed; prev reads of buf p^1 done
    if (blk + 1 < NN / 64) prefetch(blk + 1, p ^ 1);

    const half_t* kb = &Kbuf[p][0][0];
    const half_t* vb = &Vbuf[p][0][0];

    // S = Q K^T for this 16x64 block (4 subtiles x 2 K-steps), K tiles from LDS
    v8f sa[4];
#pragma unroll
    for (int t = 0; t < 4; ++t) {
#pragma unroll
      for (int i = 0; i < 8; ++i) sa[t][i] = 0.f;
#pragma unroll
      for (int s = 0; s < 2; ++s) {
        v16h kf = load_fragB(kb + (t * 16) * 64 + s * 32, 64);
        sa[t] = wmma_f16(qf[s], kf, sa[t]);
      }
    }
    // online softmax: rows 0..7 live in lanes 0..15 (slot r), rows 8..15 in lanes 16..31
#pragma unroll
    for (int r = 0; r < 8; ++r) {
      float mx = fmaxf(fmaxf(sa[0][r], sa[1][r]), fmaxf(sa[2][r], sa[3][r]));
#pragma unroll
      for (int msk = 1; msk <= 8; msk <<= 1) mx = fmaxf(mx, __shfl_xor(mx, msk, 32));
      float newm = fmaxf(m_i[r], mx);
      float corr = __expf(m_i[r] - newm);
      m_i[r] = newm;
      float ls = 0.f;
#pragma unroll
      for (int t = 0; t < 4; ++t) {
        float p2 = __expf(sa[t][r] - newm);
        Plds[warp][rowadd + r][t * 16 + colbase] = (half_t)p2;
        ls += p2;
      }
#pragma unroll
      for (int msk = 1; msk <= 8; msk <<= 1) ls += __shfl_xor(ls, msk, 32);
      l_i[r] = l_i[r] * corr + ls;
#pragma unroll
      for (int j = 0; j < 4; ++j) o[j][r] *= corr;
    }
    asm volatile("s_wait_dscnt 0" ::: "memory"); // P staged in LDS, reshape D-layout -> A-operand
    // O += P V  (2 K-steps of 32 keys x 4 d-subtiles), V tiles from LDS
#pragma unroll
    for (int s = 0; s < 2; ++s) {
      v16h pf = load_fragA(pl + s * 32, 64);
#pragma unroll
      for (int j = 0; j < 4; ++j) {
        v16h vf = load_fragB(vb + (j * 16) * 64 + s * 32, 64);
        o[j] = wmma_f16(pf, vf, o[j]);
      }
    }
  }
#pragma unroll
  for (int r = 0; r < 8; ++r) {
    float inv = 1.f / l_i[r];
    int n = q0 + rowadd + r;
#pragma unroll
    for (int j = 0; j < 4; ++j)
      Oh[((size_t)b * NN + n) * DD + j * 16 + colbase] = (half_t)(o[j][r] * inv);
  }
}

// ---------------- output projection + bias + residual: one wave -> 16(c) x 16(n) tile ----------------
__global__ __launch_bounds__(128) void k_out(const half_t* woh, const half_t* Oh, const float* bo,
                                             const float* x, float* out) {
  int wid = (blockIdx.x * 128 + threadIdx.x) >> 5;
  int nt = wid & 255;
  int ct = (wid >> 8) & 31;
  int b  = wid >> 13;
  int lane = threadIdx.x & 31;
  v8f acc;
#pragma unroll
  for (int i = 0; i < 8; ++i) acc[i] = 0.f;
#pragma unroll
  for (int s = 0; s < 2; ++s) {
    v16h a  = load_fragA(woh + ct * 16 * DD + s * 32, DD);
    v16h bb = load_fragB(Oh + ((size_t)b * NN + nt * 16) * DD + s * 32, DD);
    acc = wmma_f16(a, bb, acc);
  }
  int col = lane & 15;
  int rowadd = (lane >> 4) << 3;
  int n = nt * 16 + col;
#pragma unroll
  for (int r = 0; r < 8; ++r) {
    int c = ct * 16 + rowadd + r;
    size_t idx = ((size_t)b * CC + c) * NN + n;
    out[idx] = acc[r] + bo[c] + x[idx];
  }
}

extern "C" void kernel_launch(void* const* d_in, const int* in_sizes, int n_in,
                              void* d_out, int out_size, void* d_ws, size_t ws_size,
                              hipStream_t stream) {
  (void)in_sizes; (void)n_in; (void)out_size; (void)ws_size;
  const float* x     = (const float*)d_in[0];
  const float* gamma = (const float*)d_in[1];
  const float* beta  = (const float*)d_in[2];
  const float* wq    = (const float*)d_in[3];
  const float* bq    = (const float*)d_in[4];
  const float* wk    = (const float*)d_in[5];
  const float* bk    = (const float*)d_in[6];
  const float* wv    = (const float*)d_in[7];
  const float* bv    = (const float*)d_in[8];
  const float* wo    = (const float*)d_in[9];
  const float* bo    = (const float*)d_in[10];
  float* out = (float*)d_out;

  char* ws = (char*)d_ws;
  size_t off = 0;
  auto carve = [&](size_t bytes) -> char* {
    char* p = ws + off;
    off = (off + bytes + 255) & ~(size_t)255;
    return p;
  };
  half_t* wqh  = (half_t*)carve((size_t)DD * CC * 2);
  half_t* wkh  = (half_t*)carve((size_t)DD * CC * 2);
  half_t* wvh  = (half_t*)carve((size_t)DD * CC * 2);
  half_t* woh  = (half_t*)carve((size_t)CC * DD * 2);
  float*  mean = (float*)carve(BB * GG * 4);
  float*  rstd = (float*)carve(BB * GG * 4);
  half_t* xnT  = (half_t*)carve((size_t)BB * NN * CC * 2);
  half_t* Qh   = (half_t*)carve((size_t)BB * NN * DD * 2);
  half_t* Kh   = (half_t*)carve((size_t)BB * NN * DD * 2);
  half_t* Vt   = (half_t*)carve((size_t)BB * DD * NN * 2);
  half_t* Oh   = (half_t*)carve((size_t)BB * NN * DD * 2);

  const int WN = DD * CC; // 32768 elements per weight matrix
  k_f32_to_f16<<<(WN + 255) / 256, 256, 0, stream>>>(wq, wqh, WN);
  k_f32_to_f16<<<(WN + 255) / 256, 256, 0, stream>>>(wk, wkh, WN);
  k_f32_to_f16<<<(WN + 255) / 256, 256, 0, stream>>>(wv, wvh, WN);
  k_f32_to_f16<<<(WN + 255) / 256, 256, 0, stream>>>(wo, woh, WN);

  k_gn_stats<<<BB * GG, 256, 0, stream>>>(x, mean, rstd);
  k_gn_apply<<<BB * CC, 256, 0, stream>>>(x, gamma, beta, mean, rstd, xnT);

  // waves = B * (N/16) * (D/16) = 4*256*4 = 4096 ; 4 waves per 128-thread block
  k_proj_qk<<<4096 / 4, 128, 0, stream>>>(xnT, wqh, wkh, bq, bk, Qh, Kh);
  k_proj_v <<<4096 / 4, 128, 0, stream>>>(xnT, wvh, bv, Vt);

  // q-tiles = B * (N/16) = 1024 ; 4 per block (same batch), 256 blocks
  k_attn<<<BB * 64, 128, 0, stream>>>(Qh, Kh, Vt, Oh);

  // waves = B * (C/16) * (N/16) = 4*32*256 = 32768 ; 4 per block
  k_out<<<32768 / 4, 128, 0, stream>>>(woh, Oh, bo, x, out);
}
